// TransformerEncoder_28484223107269
// MI455X (gfx1250) — compile-verified, matrix operand order checked by reading, not run on hardware
//
#include <hip/hip_runtime.h>
#include <hip/hip_bf16.h>
#include <math.h>

typedef __attribute__((ext_vector_type(16))) _Float16 v16h;
typedef __attribute__((ext_vector_type(8)))  _Float16 v8h;
typedef __attribute__((ext_vector_type(8)))  float    v8f;

union Frag16 { v16h v; v8h p[2]; _Float16 h[16]; };

#define HW 4096
#define CCH 256
#define NB  16
#define NCH 680
#define M_FF1P 1408   // 1360 padded to mult of 64
#define K_FF2P 704    // 680 padded to mult of 32

__device__ __forceinline__ float gelu_f(float x) {
    return 0.5f * x * (1.0f + erff(x * 0.70710678118654752f));
}
__device__ __forceinline__ float softplus_f(float x) {
    return (x > 20.0f) ? x : log1pf(expf(x));
}

// ---------------- f32 -> f16 convert, zero-fill past nsrc ----------------
__global__ void cvt_pad_kernel(const float* __restrict__ in, _Float16* __restrict__ out,
                               int nsrc, int ntot) {
    int i = blockIdx.x * 256 + threadIdx.x;
    if (i < ntot) out[i] = (i < nsrc) ? (_Float16)in[i] : (_Float16)0.0f;
}

// ---------------- f32[M,Ks] -> f16[M,Kd] with zero-padded K ----------------
__global__ void cvt_padk_kernel(const float* __restrict__ in, _Float16* __restrict__ out,
                                int M, int Ks, int Kd) {
    int i = blockIdx.x * 256 + threadIdx.x;
    if (i >= M * Kd) return;
    int m = i / Kd, k = i - m * Kd;
    out[i] = (k < Ks) ? (_Float16)in[m * Ks + k] : (_Float16)0.0f;
}

// ---------------- zero the 24 pad rows of glu[b][704][4096] ----------------
__global__ void glu_pad_kernel(_Float16* __restrict__ glu) {
    size_t i = (size_t)blockIdx.x * 256 + threadIdx.x;     // NB*24*4096 total
    size_t b = i / (24 * HW);
    size_t r = i - b * (24 * HW);
    glu[b * (size_t)K_FF2P * HW + (size_t)NCH * HW + r] = (_Float16)0.0f;
}

// ---------------- InstanceNorm (affine=False) -> f16 ----------------
__global__ void __launch_bounds__(256) inorm_kernel(const float* __restrict__ in,
                                                    _Float16* __restrict__ out) {
    int bc = blockIdx.x;
    const float* p = in + (size_t)bc * HW;
    __shared__ float s1[256], s2[256];
    int tid = threadIdx.x;
    float a = 0.f, b = 0.f;
    for (int i = tid; i < HW; i += 256) { float v = p[i]; a += v; b += v * v; }
    s1[tid] = a; s2[tid] = b;
    __syncthreads();
    for (int s = 128; s > 0; s >>= 1) {
        if (tid < s) { s1[tid] += s1[tid + s]; s2[tid] += s2[tid + s]; }
        __syncthreads();
    }
    float mu  = s1[0] * (1.0f / HW);
    float var = s2[0] * (1.0f / HW) - mu * mu;
    float inv = rsqrtf(var + 1e-5f);
    for (int i = tid; i < HW; i += 256)
        out[(size_t)bc * HW + i] = (_Float16)((p[i] - mu) * inv);
}

// ---------------- WMMA GEMM: Y = A[M,K] @ X[b][K,4096] + epilogue ----------
// M % 64 == 0, K % 32 == 0 (buffers pre-padded). Software-pipelined staging.
// MODE: 0 gelu(y+bias)->f16 ; 1 softplus(y+bias)->f16 ; 2 y+bias+resid->f32
//       3 y->f16 ; 4 y+resid->f32
template <int MODE>
__global__ void __launch_bounds__(256) gemm_kernel(
    const _Float16* __restrict__ A, const _Float16* __restrict__ Xall,
    const float* __restrict__ bias, const float* __restrict__ resid,
    float* __restrict__ out32, _Float16* __restrict__ out16,
    int M, int K)
{
    const int N = HW;
    int b  = blockIdx.z;
    const _Float16* X = Xall + (size_t)b * K * N;
    int n0 = blockIdx.x * 64;
    int m0 = blockIdx.y * 64;
    int tid  = threadIdx.x;
    int lane = tid & 31, wv = tid >> 5;
    int mw = (wv & 3) * 16, nw = (wv >> 2) * 32;
    int row_l = lane & 15, kh8 = (lane >> 4) * 8;

    __shared__ __align__(16) _Float16 As[64 * 32];   // [m][k]
    __shared__ __align__(16) _Float16 Bs[64 * 32];   // transposed: [n][k]

    // staging roles
    int ar_ = tid >> 2, acb = (tid & 3) * 8;         // A: row 0..63, kcol {0,8,16,24}
    int bkk = tid >> 3, bnb = (tid & 7) * 8;         // B: k 0..31, n {0..56}
    const _Float16* Aptr = A + (size_t)(m0 + ar_) * K + acb;
    const _Float16* Xptr = X + (size_t)bkk * N + n0 + bnb;

    v8h areg = *(const v8h*)Aptr;
    v8h breg = *(const v8h*)Xptr;

    v8f c0 = {}; v8f c1 = {};
    for (int k0 = 0; k0 < K; k0 += 32) {
        *(v8h*)&As[ar_ * 32 + acb] = areg;
        #pragma unroll
        for (int i = 0; i < 8; ++i) Bs[(bnb + i) * 32 + bkk] = breg[i];
        __syncthreads();
        if (k0 + 32 < K) {   // prefetch next tile while WMMAs run
            areg = *(const v8h*)(Aptr + k0 + 32);
            breg = *(const v8h*)(Xptr + (size_t)(k0 + 32) * N);
        }
        Frag16 a, bf0, bf1;
        const _Float16* as  = &As[(mw + row_l) * 32 + kh8];
        const _Float16* b0s = &Bs[(nw + row_l) * 32 + kh8];
        const _Float16* b1s = &Bs[(nw + 16 + row_l) * 32 + kh8];
        a.p[0]   = *(const v8h*)as;   a.p[1]   = *(const v8h*)(as + 16);
        bf0.p[0] = *(const v8h*)b0s;  bf0.p[1] = *(const v8h*)(b0s + 16);
        bf1.p[0] = *(const v8h*)b1s;  bf1.p[1] = *(const v8h*)(b1s + 16);
        c0 = __builtin_amdgcn_wmma_f32_16x16x32_f16(false, a.v, false, bf0.v, (short)0, c0, false, false);
        c1 = __builtin_amdgcn_wmma_f32_16x16x32_f16(false, a.v, false, bf1.v, (short)0, c1, false, false);
        __syncthreads();
    }

    size_t obase = (size_t)b * M * N;
    #pragma unroll
    for (int r = 0; r < 8; ++r) {
        int gm = m0 + mw + r + ((lane >> 4) << 3);
        size_t rowo = obase + (size_t)gm * N;
        int gn0 = n0 + nw + (lane & 15);
        float bb = (MODE <= 2) ? bias[gm] : 0.0f;
        #pragma unroll
        for (int e = 0; e < 2; ++e) {
            float vvv = (e ? c1[r] : c0[r]) + bb;
            int gn = gn0 + (e ? 16 : 0);
            if (MODE == 0)      out16[rowo + gn] = (_Float16)gelu_f(vvv);
            else if (MODE == 1) out16[rowo + gn] = (_Float16)softplus_f(vvv);
            else if (MODE == 2) out32[rowo + gn] = vvv + resid[rowo + gn];
            else if (MODE == 3) out16[rowo + gn] = (_Float16)vvv;
            else                out32[rowo + gn] = vvv + resid[rowo + gn];
        }
    }
}

// ---------------- ksum[b*256+ch] = sum_n k16[b,ch,n] ----------------
__global__ void __launch_bounds__(256) ksum_kernel(const _Float16* __restrict__ k16,
                                                   float* __restrict__ ksum) {
    int bc = blockIdx.x;
    const _Float16* p = k16 + (size_t)bc * HW;
    __shared__ float s1[256];
    int tid = threadIdx.x;
    float a = 0.f;
    for (int i = tid; i < HW; i += 256) a += (float)p[i];
    s1[tid] = a;
    __syncthreads();
    for (int s = 128; s > 0; s >>= 1) {
        if (tid < s) s1[tid] += s1[tid + s];
        __syncthreads();
    }
    if (tid == 0) ksum[bc] = s1[0];
}

// ---------------- kv[bh][c][d] = sum_n k[c,n]*v[d,n] (64x64, K=4096) --------
__global__ void __launch_bounds__(256) kv_kernel(const _Float16* __restrict__ k16,
                                                 const _Float16* __restrict__ v16,
                                                 float* __restrict__ kv) {
    int bh = blockIdx.x;
    int b = bh >> 2, h = bh & 3;
    const _Float16* kp = k16 + ((size_t)b * CCH + h * 64) * HW;
    const _Float16* vp = v16 + ((size_t)b * CCH + h * 64) * HW;
    int tid = threadIdx.x;
    int lane = tid & 31, wv = tid >> 5;
    int mw = (wv & 3) * 16, nw = (wv >> 2) * 32;
    int row_l = lane & 15, kh8 = (lane >> 4) * 8;

    const _Float16* ar  = kp + (size_t)(mw + row_l) * HW + kh8;
    const _Float16* br0 = vp + (size_t)(nw + row_l) * HW + kh8;
    const _Float16* br1 = vp + (size_t)(nw + 16 + row_l) * HW + kh8;

    v8f c0 = {}; v8f c1 = {};
    for (int k0 = 0; k0 < HW; k0 += 32) {
        Frag16 a, bf0, bf1;
        a.p[0]   = *(const v8h*)(ar + k0);        a.p[1]   = *(const v8h*)(ar + k0 + 16);
        bf0.p[0] = *(const v8h*)(br0 + k0);       bf0.p[1] = *(const v8h*)(br0 + k0 + 16);
        bf1.p[0] = *(const v8h*)(br1 + k0);       bf1.p[1] = *(const v8h*)(br1 + k0 + 16);
        c0 = __builtin_amdgcn_wmma_f32_16x16x32_f16(false, a.v, false, bf0.v, (short)0, c0, false, false);
        c1 = __builtin_amdgcn_wmma_f32_16x16x32_f16(false, a.v, false, bf1.v, (short)0, c1, false, false);
    }
    float* kvb = kv + (size_t)bh * 4096;
    #pragma unroll
    for (int r = 0; r < 8; ++r) {
        int cc = mw + r + ((lane >> 4) << 3);
        int d0 = nw + (lane & 15);
        kvb[cc * 64 + d0]      = c0[r];
        kvb[cc * 64 + d0 + 16] = c1[r];
    }
}

// ---------------- z[bh][n] = 1 / (dot(q[:,n], ksum)/8 + 4096) ----------------
__global__ void __launch_bounds__(256) z_kernel(const _Float16* __restrict__ q16,
                                                const float* __restrict__ ksum,
                                                float* __restrict__ z) {
    size_t g = (size_t)blockIdx.x * 256 + threadIdx.x;  // B*4*4096 total
    int n  = (int)(g & 4095);
    int bh = (int)(g >> 12);
    int b = bh >> 2, h = bh & 3;
    const _Float16* qp = q16 + ((size_t)b * CCH + h * 64) * HW + n;
    const float* ks = ksum + b * CCH + h * 64;
    float acc = 0.f;
    #pragma unroll 8
    for (int c = 0; c < 64; ++c) acc += (float)qp[(size_t)c * HW] * ks[c];
    z[g] = 1.0f / (acc * 0.125f + 4096.0f);
}

// ---- out16[b, h*64+d, n] = ((kv^T @ q)/8 + v) * z * g  ----
__global__ void __launch_bounds__(256) attn_out_kernel(
    const _Float16* __restrict__ q16, const _Float16* __restrict__ v16,
    const _Float16* __restrict__ g16, const float* __restrict__ kv,
    const float* __restrict__ z, _Float16* __restrict__ out16) {
    int n0 = blockIdx.x * 64;
    int bh = blockIdx.y;
    int b = bh >> 2, h = bh & 3;
    int tid = threadIdx.x;

    __shared__ __align__(16) _Float16 Akv[64 * 64];  // A[d][c] = kv[c][d]
    __shared__ __align__(16) _Float16 Bq[64 * 64];   // [n][c]

    const float* kvp = kv + (size_t)bh * 4096;
    for (int idx = tid; idx < 4096; idx += 256) {
        int d = idx >> 6, c = idx & 63;
        Akv[d * 64 + c] = (_Float16)kvp[c * 64 + d];
    }
    const _Float16* qp = q16 + ((size_t)b * CCH + h * 64) * HW;
    {
        int cc = tid >> 3, nb = (tid & 7) * 8;
        for (int c2 = cc; c2 < 64; c2 += 32) {
            v8h qv = *(const v8h*)(qp + (size_t)c2 * HW + n0 + nb);
            #pragma unroll
            for (int i = 0; i < 8; ++i) Bq[(nb + i) * 64 + c2] = qv[i];
        }
    }
    __syncthreads();

    int lane = tid & 31, wv = tid >> 5;
    int mw = (wv & 3) * 16, nw = (wv >> 2) * 32;
    int row_l = lane & 15, kh8 = (lane >> 4) * 8;
    v8f c0 = {}; v8f c1 = {};
    #pragma unroll
    for (int k0 = 0; k0 < 64; k0 += 32) {
        Frag16 a, bf0, bf1;
        const _Float16* as  = &Akv[(mw + row_l) * 64 + k0 + kh8];
        const _Float16* b0s = &Bq[(nw + row_l) * 64 + k0 + kh8];
        const _Float16* b1s = &Bq[(nw + 16 + row_l) * 64 + k0 + kh8];
        a.p[0]   = *(const v8h*)as;   a.p[1]   = *(const v8h*)(as + 16);
        bf0.p[0] = *(const v8h*)b0s;  bf0.p[1] = *(const v8h*)(b0s + 16);
        bf1.p[0] = *(const v8h*)b1s;  bf1.p[1] = *(const v8h*)(b1s + 16);
        c0 = __builtin_amdgcn_wmma_f32_16x16x32_f16(false, a.v, false, bf0.v, (short)0, c0, false, false);
        c1 = __builtin_amdgcn_wmma_f32_16x16x32_f16(false, a.v, false, bf1.v, (short)0, c1, false, false);
    }

    const _Float16* vp = v16 + ((size_t)b * CCH + h * 64) * HW;
    const _Float16* gp = g16 + ((size_t)b * CCH + h * 64) * HW;
    const float* zp = z + (size_t)bh * HW;
    #pragma unroll
    for (int r = 0; r < 8; ++r) {
        int d = mw + r + ((lane >> 4) << 3);
        size_t rowo = ((size_t)b * CCH + h * 64 + d) * HW;
        #pragma unroll
        for (int e = 0; e < 2; ++e) {
            int n = n0 + nw + (lane & 15) + (e ? 16 : 0);
            float val = (e ? c1[r] : c0[r]) * 0.125f + (float)vp[(size_t)d * HW + n];
            val *= zp[n];
            val *= (float)gp[(size_t)d * HW + n];
            out16[rowo + n] = (_Float16)val;
        }
    }
}

// ---- depthwise 3x3 (SAME) on y1[b,1408(stride),64,64] then glu = gelu(x1)*x2
__global__ void __launch_bounds__(256) dwglu_kernel(const _Float16* __restrict__ y1,
                                                    const float* __restrict__ wd,
                                                    _Float16* __restrict__ glu) {
    int n = blockIdx.x * 256 + threadIdx.x;  // 0..4095
    int c = blockIdx.y;                      // 0..679
    int b = blockIdx.z;
    int x = n & 63, y = n >> 6;
    const _Float16* p1 = y1 + ((size_t)b * M_FF1P + c) * HW;
    const _Float16* p2 = y1 + ((size_t)b * M_FF1P + c + NCH) * HW;
    const float* w1 = wd + (size_t)c * 9;
    const float* w2 = wd + (size_t)(c + NCH) * 9;
    float a1 = 0.f, a2 = 0.f;
    #pragma unroll
    for (int dy = -1; dy <= 1; ++dy) {
        int yy = y + dy;
        if (yy < 0 || yy > 63) continue;
        #pragma unroll
        for (int dx = -1; dx <= 1; ++dx) {
            int xx = x + dx;
            if (xx < 0 || xx > 63) continue;
            int off = yy * 64 + xx;
            int wi = (dy + 1) * 3 + (dx + 1);
            a1 += w1[wi] * (float)p1[off];
            a2 += w2[wi] * (float)p2[off];
        }
    }
    glu[((size_t)b * K_FF2P + c) * HW + n] = (_Float16)(gelu_f(a1) * a2);
}

extern "C" void kernel_launch(void* const* d_in, const int* in_sizes, int n_in,
                              void* d_out, int out_size, void* d_ws, size_t ws_size,
                              hipStream_t stream) {
    (void)in_sizes; (void)n_in; (void)out_size; (void)ws_size;

    const float* x = (const float*)d_in[0];
    const float* W[7];  for (int i = 0; i < 7; ++i) W[i]   = (const float*)d_in[1 + i];
    const float* Bi[7]; for (int i = 0; i < 7; ++i) Bi[i]  = (const float*)d_in[8 + i];
    const float* ffw1 = (const float*)d_in[15];
    const float* ffwd = (const float*)d_in[16];
    const float* ffw2 = (const float*)d_in[17];
    float* out = (float*)d_out;

    const size_t BCHW = (size_t)NB * CCH * HW;
    char* ws = (char*)d_ws;
    size_t off = 0;
    auto bump = [&](size_t bytes) -> char* {
        char* p = ws + off;
        off = (off + bytes + 255) & ~(size_t)255;
        return p;
    };

    _Float16* w16[7];
    for (int i = 0; i < 7; ++i) w16[i] = (_Float16*)bump(65536 * 2);
    _Float16* ffw1_16 = (_Float16*)bump((size_t)M_FF1P * 256 * 2);     // padded rows
    _Float16* ffw2_16 = (_Float16*)bump((size_t)256 * K_FF2P * 2);     // padded cols
    float* kv   = (float*)bump((size_t)NB * 4 * 64 * 64 * 4);
    float* ksum = (float*)bump((size_t)NB * CCH * 4);
    float* zb   = (float*)bump((size_t)NB * 4 * HW * 4);
    float* attn = (float*)bump(BCHW * 4);
    _Float16* xn  = (_Float16*)bump(BCHW * 2);
    _Float16* tmp = (_Float16*)bump(BCHW * 2);
    _Float16* q16 = (_Float16*)bump(BCHW * 2);
    _Float16* k16 = (_Float16*)bump(BCHW * 2);
    _Float16* v16 = (_Float16*)bump(BCHW * 2);
    _Float16* g16 = (_Float16*)bump(BCHW * 2);
    // phase-2 reuse: y1 (16*1408*4096 f16 = 176 MiB) overlays xn..g (192 MiB),
    // all of which are dead once attn is formed.
    _Float16* y1  = xn;
    _Float16* glu = (_Float16*)bump((size_t)NB * K_FF2P * HW * 2);
    _Float16* yn  = (_Float16*)bump(BCHW * 2);

    // ---- weight conversion (with padding) ----
    for (int i = 0; i < 7; ++i)
        cvt_pad_kernel<<<(65536 + 255) / 256, 256, 0, stream>>>(W[i], w16[i], 65536, 65536);
    cvt_pad_kernel<<<(M_FF1P * 256 + 255) / 256, 256, 0, stream>>>(
        ffw1, ffw1_16, 2 * NCH * 256, M_FF1P * 256);
    cvt_padk_kernel<<<(256 * K_FF2P + 255) / 256, 256, 0, stream>>>(
        ffw2, ffw2_16, 256, NCH, K_FF2P);
    glu_pad_kernel<<<(NB * 24 * HW) / 256, 256, 0, stream>>>(glu);

    // ---- mGAttn ----
    inorm_kernel<<<NB * CCH, 256, 0, stream>>>(x, xn);

    dim3 gC(64, 4, NB);  // N-tiles, M-tiles (256/64), batch
    gemm_kernel<0><<<gC, 256, 0, stream>>>(w16[0], xn,  Bi[0], nullptr, nullptr, tmp, 256, 256);
    gemm_kernel<1><<<gC, 256, 0, stream>>>(w16[1], tmp, Bi[1], nullptr, nullptr, q16, 256, 256);
    gemm_kernel<0><<<gC, 256, 0, stream>>>(w16[2], xn,  Bi[2], nullptr, nullptr, tmp, 256, 256);
    gemm_kernel<1><<<gC, 256, 0, stream>>>(w16[3], tmp, Bi[3], nullptr, nullptr, k16, 256, 256);
    gemm_kernel<0><<<gC, 256, 0, stream>>>(w16[4], xn,  Bi[4], nullptr, nullptr, v16, 256, 256);
    gemm_kernel<0><<<gC, 256, 0, stream>>>(w16[5], xn,  Bi[5], nullptr, nullptr, g16, 256, 256);

    ksum_kernel<<<NB * CCH, 256, 0, stream>>>(k16, ksum);
    kv_kernel<<<NB * 4, 256, 0, stream>>>(k16, v16, kv);
    z_kernel<<<(NB * 4 * HW) / 256, 256, 0, stream>>>(q16, ksum, zb);
    attn_out_kernel<<<dim3(64, NB * 4), 256, 0, stream>>>(q16, v16, g16, kv, zb, tmp);

    // attn = wo @ out + bo + x  (f32)
    gemm_kernel<2><<<gC, 256, 0, stream>>>(w16[6], tmp, Bi[6], x, attn, nullptr, 256, 256);

    // ---- FeedForward ----
    inorm_kernel<<<NB * CCH, 256, 0, stream>>>(attn, yn);

    dim3 gF1(64, M_FF1P / 64, NB);  // 22 M-tiles, exact
    gemm_kernel<3><<<gF1, 256, 0, stream>>>(ffw1_16, yn, nullptr, nullptr, nullptr, y1, M_FF1P, 256);

    dwglu_kernel<<<dim3(HW / 256, NCH, NB), 256, 0, stream>>>(y1, ffwd, glu);

    // out = ff_w2 @ glu + attn  (K padded to 704, pad rows of glu are zero)
    gemm_kernel<4><<<gC, 256, 0, stream>>>(ffw2_16, glu, nullptr, attn, out, nullptr, 256, K_FF2P);
}
